// LMMD_loss_65541200937312
// MI455X (gfx1250) — matrix operand order repaired and checked by reading, not compiled, not run on hardware
//
#include <hip/hip_runtime.h>

// ---------------------------------------------------------------------------
// LMMD loss for MI455X (gfx1250).
//   loss = inv_len * sum_{i,j} (b_i . b_j) * K[i,j],  i,j over 8192 rows
//   K = sum_{t=0..4} exp(-L2/(bw*2^t)) = E + sqrt(E) + ... (E=exp(-L2/bw))
//   L2 = max(sq_i + sq_j - 2*Gram_ij, 0),  Gram via split-bf16 WMMA (3x)
//   bw  from closed form sumL2 = 2n*sum(sq) - 2*||colsum||^2  (no 67M pass)
// Everything (x: 16MB split-bf16, b: 1MB) is L2-resident -> compute bound.
// ---------------------------------------------------------------------------

typedef __attribute__((ext_vector_type(16))) __bf16 v16bf;
typedef __attribute__((ext_vector_type(8)))  float  v8f;

#define N_TOT 8192
#define BROWS 4096
#define DIMS  512
#define CLS   31

union FragB { v16bf v; uint4 q[2]; };

static __device__ inline v8f wmma_bf16(v16bf a, v16bf b, v8f c) {
  // D = A(16x32) * B(32x16) + C, fp32 accumulate
  return __builtin_amdgcn_wmma_f32_16x16x32_bf16(false, a, false, b,
                                                 (short)0, c, false, false);
}

static __device__ inline unsigned short f2bf(float v) {  // RNE truncation
  unsigned u = __float_as_uint(v);
  u += 0x7FFFu + ((u >> 16) & 1u);
  return (unsigned short)(u >> 16);
}
static __device__ inline float bf2f(unsigned short h) {
  return __uint_as_float(((unsigned)h) << 16);
}

// ---------------- workspace layout (bytes, all 256-aligned) ----------------
#define OFF_XHI   ((size_t)0)          //  8192*512*2 = 8 MiB   x high bf16
#define OFF_XLO   ((size_t)8388608)    //  8 MiB                x low  bf16
#define OFF_WBH   ((size_t)16777216)   //  8192*32*2 = 512 KiB  b high bf16
#define OFF_WBL   ((size_t)17301504)   //  512 KiB              b low  bf16
#define OFF_SQ    ((size_t)17825792)   //  8192*4               row sq norms
#define OFF_COLP  ((size_t)17858560)   //  512*512*4 = 1 MiB    colsum partials
#define OFF_SQP   ((size_t)18907136)   //  512*4                sumsq partials
#define OFF_RS    ((size_t)18909184)   //  32*4                 masked 1/s_sum
#define OFF_RT    ((size_t)18909440)   //  32*4                 masked 1/t_sum
#define OFF_SCAL  ((size_t)18909696)   //  scal[0]=-1/bw, scal[1]=inv_len
#define OFF_NAN   ((size_t)18909952)   //  uint flag
#define OFF_PART  ((size_t)18910208)   //  8192*4               block partials

// ---------------------------------------------------------------------------
// Pass 1: fp32 -> split bf16 (hi/lo), per-row sq norms, column-sum partials.
// 64 blocks x 256 thr; wave handles 16 rows, lane owns cols {lane+32t}.
// ---------------------------------------------------------------------------
__global__ __launch_bounds__(256) void prep_x(
    const float* __restrict__ src, const float* __restrict__ tgt,
    unsigned short* __restrict__ Xhi, unsigned short* __restrict__ Xlo,
    float* __restrict__ sq, float* __restrict__ colpart,
    float* __restrict__ sqpart, unsigned* __restrict__ nanflag)
{
  const int tid = threadIdx.x;
  const int wave = tid >> 5, lane = tid & 31;
  const int rowbase = blockIdx.x * 128 + wave * 16;
  float csum[16];
#pragma unroll
  for (int t = 0; t < 16; ++t) csum[t] = 0.f;
  float wavesq = 0.f;
  for (int rr = 0; rr < 16; ++rr) {
    const int row = rowbase + rr;
    const float* xr = (row < BROWS) ? (src + (size_t)row * DIMS)
                                    : (tgt + (size_t)(row - BROWS) * DIMS);
    float s = 0.f;
#pragma unroll
    for (int t = 0; t < 16; ++t) {
      const int col = lane + 32 * t;
      const float v = xr[col];
      csum[t] += v;
      s = fmaf(v, v, s);
      const unsigned short h = f2bf(v);
      Xhi[(size_t)row * DIMS + col] = h;
      Xlo[(size_t)row * DIMS + col] = f2bf(v - bf2f(h));
    }
#pragma unroll
    for (int off = 16; off > 0; off >>= 1) s += __shfl_xor(s, off, 32);
    if (lane == 0) sq[row] = s;
    wavesq += s;                       // same value in all lanes after butterfly
  }
  const int wid = blockIdx.x * 8 + wave;
#pragma unroll
  for (int t = 0; t < 16; ++t)
    colpart[(size_t)wid * DIMS + lane + 32 * t] = csum[t];
  if (lane == 0) sqpart[wid] = wavesq;
  if (blockIdx.x == 0 && tid == 0) *nanflag = 0u;
}

// ---------------------------------------------------------------------------
// Pass 2 (1 block): class statistics, mask, inv_len, bandwidth coefficient.
// All reductions fixed-order (integer LDS atomics + deterministic trees).
// ---------------------------------------------------------------------------
__global__ __launch_bounds__(256) void prep_labels(
    const int* __restrict__ s_label, const float* __restrict__ t_label,
    const float* __restrict__ colpart, const float* __restrict__ sqpart,
    float* __restrict__ rs, float* __restrict__ rt, float* __restrict__ scal)
{
  __shared__ int   s_cnt[CLS];
  __shared__ int   tp[CLS];
  __shared__ float tpart[CLS * 8];
  __shared__ float red[256];
  __shared__ float red2[256];
  const int tid = threadIdx.x;
  if (tid < CLS) { s_cnt[tid] = 0; tp[tid] = 0; }
  __syncthreads();
  for (int i = tid; i < BROWS; i += 256) atomicAdd(&s_cnt[s_label[i]], 1);
  for (int i = tid; i < BROWS; i += 256) {
    const float* tr = t_label + (size_t)i * CLS;
    float best = tr[0]; int arg = 0;
    for (int c = 1; c < CLS; ++c) { float v = tr[c]; if (v > best) { best = v; arg = c; } }
    atomicOr(&tp[arg], 1);             // argmax presence (first-max, like jnp)
  }
  if (tid < CLS * 8) {                 // t_label column-sum partials, fixed order
    const int c = tid >> 3, ch = tid & 7;
    float s = 0.f;
    for (int i = ch * 512; i < (ch + 1) * 512; ++i) s += t_label[(size_t)i * CLS + c];
    tpart[tid] = s;
  }
  // closed-form sum(L2): 2n*sumsq - 2*||colsum||^2
  float cs0 = 0.f, cs1 = 0.f;
  for (int p = 0; p < 512; ++p) {
    cs0 += colpart[(size_t)p * DIMS + tid];
    cs1 += colpart[(size_t)p * DIMS + tid + 256];
  }
  red[tid]  = cs0 * cs0 + cs1 * cs1;
  red2[tid] = sqpart[tid] + sqpart[tid + 256];
  __syncthreads();
  for (int off = 128; off > 0; off >>= 1) {
    if (tid < off) { red[tid] += red[tid + off]; red2[tid] += red2[tid + off]; }
    __syncthreads();
  }
  if (tid == 0) {
    const float sumL2 = 2.f * (8192.f * red2[0] - red[0]);
    const float bw    = (sumL2 / 67100672.f) * 0.25f;   // /(n^2-n) / 2^(5/2 floor)
    scal[0] = -1.f / bw;
    int len = 0;
    for (int c = 0; c < CLS; ++c) if (s_cnt[c] > 0 && tp[c]) ++len;
    scal[1] = (len > 0) ? 1.f / (float)len : 0.f;
  }
  if (tid < 32) {
    float rsv = 0.f, rtv = 0.f;
    if (tid < CLS && s_cnt[tid] > 0 && tp[tid]) {
      rsv = 1.f / (float)s_cnt[tid];
      float ts = 0.f;
      for (int ch = 0; ch < 8; ++ch) ts += tpart[tid * 8 + ch];
      rtv = 1.f / ((ts == 0.f) ? 1.f : ts);
    }
    rs[tid] = rsv; rt[tid] = rtv;
  }
}

// ---------------------------------------------------------------------------
// Pass 3: build signed weight vectors b[8192][32] (target rows negated,
// col 31 zero-padded), stored as split bf16 for the W-tile WMMA.
// ---------------------------------------------------------------------------
__global__ __launch_bounds__(256) void prep_b(
    const int* __restrict__ s_label, const float* __restrict__ t_label,
    const float* __restrict__ rs, const float* __restrict__ rt,
    unsigned short* __restrict__ Wbh, unsigned short* __restrict__ Wbl)
{
  const int i = blockIdx.x * 256 + threadIdx.x;
  unsigned short* oh = Wbh + (size_t)i * 32;
  unsigned short* ol = Wbl + (size_t)i * 32;
  if (i < BROWS) {
    const int lab = s_label[i];
    const float v0 = rs[lab];
    for (int c = 0; c < 32; ++c) {
      const float v = (c == lab) ? v0 : 0.f;
      const unsigned short h = f2bf(v);
      oh[c] = h; ol[c] = f2bf(v - bf2f(h));
    }
  } else {
    const float* tr = t_label + (size_t)(i - BROWS) * CLS;
    for (int c = 0; c < 32; ++c) {
      const float v = (c < CLS) ? (-tr[c] * rt[c]) : 0.f;
      const unsigned short h = f2bf(v);
      oh[c] = h; ol[c] = f2bf(v - bf2f(h));
    }
  }
}

// ---------------------------------------------------------------------------
// Main fused pass: per wave a 16x64 strip of the 8192x8192 kernel matrix.
// Gram via 3x split-bf16 WMMA per 32-K step; W tile via 3 WMMAs (K=32 pad);
// then L2 -> exp + sqrt-chain -> weighted FMA, all in registers.
//   A frag layout: lane m=l%16, half=l/16; cols {k+8h..+7} U {k+16+8h..+7}
//   B frag layout: lane n=l%16, half=l/16; cols {k+8h..+7} U {k+16+8h..+7}
//   C/D layout   : elem r -> (row r+8*half, col l%16)
// ---------------------------------------------------------------------------
__global__ __launch_bounds__(256) void lmmd_main(
    const unsigned short* __restrict__ Xhi, const unsigned short* __restrict__ Xlo,
    const unsigned short* __restrict__ Wbh, const unsigned short* __restrict__ Wbl,
    const float* __restrict__ sq, const float* __restrict__ scal,
    float* __restrict__ partials, unsigned* __restrict__ nanflag)
{
  const int tid  = threadIdx.x;
  const int wave = tid >> 5, lane = tid & 31;
  const int m    = lane & 15, half = lane >> 4;
  const int bi   = (int)blockIdx.x >> 7;     // 64 row-blocks of 128
  const int bj   = (int)blockIdx.x & 127;    // 128 col-blocks of 64
  const int i0   = bi * 128 + wave * 16;
  const int j0   = bj * 64;
  const float coef = scal[0];                // -1/bw

  v8f acc[4];
#pragma unroll
  for (int t = 0; t < 4; ++t)
#pragma unroll
    for (int r = 0; r < 8; ++r) acc[t][r] = 0.0f;

  const unsigned short* arh = Xhi + (size_t)(i0 + m) * DIMS;
  const unsigned short* arl = Xlo + (size_t)(i0 + m) * DIMS;
  for (int k = 0; k < DIMS; k += 32) {
    FragB ah, al;
    const int c0 = k + 8 * half;
    ah.q[0] = *(const uint4*)(arh + c0);
    ah.q[1] = *(const uint4*)(arh + c0 + 16);
    al.q[0] = *(const uint4*)(arl + c0);
    al.q[1] = *(const uint4*)(arl + c0 + 16);
#pragma unroll
    for (int t = 0; t < 4; ++t) {
      const int brow = j0 + 16 * t + m;
      const unsigned short* bph = Xhi + (size_t)brow * DIMS + k + 8 * half;
      const unsigned short* bpl = Xlo + (size_t)brow * DIMS + k + 8 * half;
      FragB bh, bl;
      bh.q[0] = *(const uint4*)(bph);
      bh.q[1] = *(const uint4*)(bph + 16);
      bl.q[0] = *(const uint4*)(bpl);
      bl.q[1] = *(const uint4*)(bpl + 16);
      acc[t] = wmma_bf16(ah.v, bh.v, acc[t]);   // hi*hi
      acc[t] = wmma_bf16(ah.v, bl.v, acc[t]);   // hi*lo
      acc[t] = wmma_bf16(al.v, bh.v, acc[t]);   // lo*hi  (~fp32 Gram)
    }
  }

  float sqi[8];
#pragma unroll
  for (int r = 0; r < 8; ++r) sqi[r] = sq[i0 + 8 * half + r];

  FragB wah, wal;                               // weight A frag (K=0..31)
  {
    const unsigned short* p  = Wbh + (size_t)(i0 + m) * 32 + 8 * half;
    const unsigned short* pl = Wbl + (size_t)(i0 + m) * 32 + 8 * half;
    wah.q[0] = *(const uint4*)(p);  wah.q[1] = *(const uint4*)(p + 16);
    wal.q[0] = *(const uint4*)(pl); wal.q[1] = *(const uint4*)(pl + 16);
  }

  float wsum = 0.f;
  unsigned nf = 0u;
#pragma unroll
  for (int t = 0; t < 4; ++t) {
    const int brow = j0 + 16 * t + m;
    FragB wbh, wbl;
    const unsigned short* p  = Wbh + (size_t)brow * 32 + 8 * half;
    const unsigned short* pl = Wbl + (size_t)brow * 32 + 8 * half;
    wbh.q[0] = *(const uint4*)(p);  wbh.q[1] = *(const uint4*)(p + 16);
    wbl.q[0] = *(const uint4*)(pl); wbl.q[1] = *(const uint4*)(pl + 16);
    v8f w;
#pragma unroll
    for (int r = 0; r < 8; ++r) w[r] = 0.f;
    w = wmma_bf16(wah.v, wbh.v, w);
    w = wmma_bf16(wah.v, wbl.v, w);
    w = wmma_bf16(wal.v, wbh.v, w);

    const float sqj = sq[brow];
#pragma unroll
    for (int r = 0; r < 8; ++r) {
      float l2 = fmaxf(sqi[r] + sqj - 2.0f * acc[t][r], 0.0f);
      float e0 = __expf(l2 * coef);            // exp(-L2/bw)
      float e1 = sqrtf(e0);                    // exp(-L2/(2bw))
      float e2 = sqrtf(e1);
      float e3 = sqrtf(e2);
      float e4 = sqrtf(e3);
      float kk = e0 + e1 + e2 + e3 + e4;
      if (!(kk == kk)) nf = 1u;
      wsum = fmaf(w[r], kk, wsum);
    }
  }

#pragma unroll
  for (int off = 16; off > 0; off >>= 1) {
    wsum += __shfl_xor(wsum, off, 32);
    nf   |= (unsigned)__shfl_xor((int)nf, off, 32);
  }
  __shared__ float    wred[8];
  __shared__ unsigned nred[8];
  if (lane == 0) { wred[wave] = wsum; nred[wave] = nf; }
  __syncthreads();
  if (tid == 0) {
    float s = 0.f; unsigned f = 0u;
    for (int wv = 0; wv < 8; ++wv) { s += wred[wv]; f |= nred[wv]; }
    partials[blockIdx.x] = s;
    if (f) atomicOr(nanflag, 1u);
  }
}

// ---------------------------------------------------------------------------
// Final fixed-order reduction of the 8192 block partials.
// ---------------------------------------------------------------------------
__global__ __launch_bounds__(256) void lmmd_finalize(
    const float* __restrict__ partials, const float* __restrict__ scal,
    const unsigned* __restrict__ nanflag, float* __restrict__ out)
{
  __shared__ float red[256];
  const int tid = threadIdx.x;
  float s = 0.f;
  for (int i = tid; i < 8192; i += 256) s += partials[i];
  red[tid] = s;
  __syncthreads();
  for (int off = 128; off > 0; off >>= 1) {
    if (tid < off) red[tid] += red[tid + off];
    __syncthreads();
  }
  if (tid == 0) out[0] = (*nanflag) ? 0.f : red[0] * scal[1];
}

// ---------------------------------------------------------------------------
extern "C" void kernel_launch(void* const* d_in, const int* in_sizes, int n_in,
                              void* d_out, int out_size, void* d_ws, size_t ws_size,
                              hipStream_t stream) {
  (void)in_sizes; (void)n_in; (void)out_size; (void)ws_size;
  const float* src     = (const float*)d_in[0];
  const float* tgt     = (const float*)d_in[1];
  const int*   s_label = (const int*)d_in[2];
  const float* t_label = (const float*)d_in[3];

  char* ws = (char*)d_ws;
  unsigned short* Xhi = (unsigned short*)(ws + OFF_XHI);
  unsigned short* Xlo = (unsigned short*)(ws + OFF_XLO);
  unsigned short* Wbh = (unsigned short*)(ws + OFF_WBH);
  unsigned short* Wbl = (unsigned short*)(ws + OFF_WBL);
  float* sq      = (float*)(ws + OFF_SQ);
  float* colpart = (float*)(ws + OFF_COLP);
  float* sqpart  = (float*)(ws + OFF_SQP);
  float* rs      = (float*)(ws + OFF_RS);
  float* rt      = (float*)(ws + OFF_RT);
  float* scal    = (float*)(ws + OFF_SCAL);
  unsigned* nanf = (unsigned*)(ws + OFF_NAN);
  float* partials= (float*)(ws + OFF_PART);

  prep_x     <<<64,   256, 0, stream>>>(src, tgt, Xhi, Xlo, sq, colpart, sqpart, nanf);
  prep_labels<<<1,    256, 0, stream>>>(s_label, t_label, colpart, sqpart, rs, rt, scal);
  prep_b     <<<32,   256, 0, stream>>>(s_label, t_label, rs, rt, Wbh, Wbl);
  lmmd_main  <<<8192, 256, 0, stream>>>(Xhi, Xlo, Wbh, Wbl, sq, scal, partials, nanf);
  lmmd_finalize<<<1,  256, 0, stream>>>(partials, scal, nanf, (float*)d_out);
}
